// Graph_Attention_1803886264721
// MI455X (gfx1250) — matrix operand order repaired and checked by reading, not compiled, not run on hardware
//
#include <hip/hip_runtime.h>
#include <hip/hip_bf16.h>

#define BB 2
#define NN 2048
#define DD 128
#define HH 8
#define OUTD 256

typedef __bf16 bf16x16 __attribute__((ext_vector_type(16)));
typedef float f32x8 __attribute__((ext_vector_type(8)));

union Frag16 {
    bf16x16 v;
    unsigned short u[16];
    uint4 q[2];
};

__device__ inline unsigned short f2bf(float f) {
    union { float f; unsigned u; } a; a.f = f;
    unsigned r = a.u + 0x7FFFu + ((a.u >> 16) & 1u);   // round-to-nearest-even
    return (unsigned short)(r >> 16);
}

__device__ inline f32x8 zero8() { f32x8 z; for (int i = 0; i < 8; ++i) z[i] = 0.0f; return z; }

__device__ inline f32x8 wmma_bf16(const Frag16& a, const Frag16& b, f32x8 c) {
    return __builtin_amdgcn_wmma_f32_16x16x32_bf16(false, a.v, false, b.v, (short)0, c, false, false);
}

// A-matrix 16x32 bf16 fragment from row-major LDS tile (row stride ld, K offset folded into base).
// Lane: row M = lane&15, half = lane>>4. Element e -> K = 16*(e>>3) + 8*half + (e&7).
__device__ inline Frag16 load_afrag(const unsigned short* base, int ld, int lane) {
    int half = lane >> 4, m = lane & 15;
    Frag16 f;
    const unsigned short* p = base + m * ld + 8 * half;
    f.q[0] = *(const uint4*)p;
    f.q[1] = *(const uint4*)(p + 16);
    return f;
}

// B-matrix 32x16 bf16 fragment; storage is N-major: element (K,N) at base[N*ld + K].
// Lane: col N = lane&15, half = lane>>4. Element e -> K = 16*(e>>3) + 8*half + (e&7).
__device__ inline Frag16 load_bfrag(const unsigned short* base, int ld, int lane) {
    int half = lane >> 4, m = lane & 15;
    Frag16 f;
    const unsigned short* p = base + m * ld + 8 * half;
    f.q[0] = *(const uint4*)p;
    f.q[1] = *(const uint4*)(p + 16);
    return f;
}

// ---------------- Kernel 1: adjacency reduction (the 134 MB streaming pass) ----------------
__global__ __launch_bounds__(256) void adjsum_kernel(const float4* __restrict__ adj,
                                                     float* __restrict__ asum, long long n) {
    long long i = (long long)blockIdx.x * 256 + threadIdx.x;
    if (i < n) { float4 a = adj[i]; asum[i] = a.x + a.y + a.z + a.w; }
}

// ---------------- Kernel 2: fused QKV projection (bf16 out), WMMA bf16 ----------------
__global__ __launch_bounds__(256) void qkv_kernel(
    const float* __restrict__ X,
    const float* __restrict__ Wq, const float* __restrict__ bq,
    const float* __restrict__ Wk, const float* __restrict__ bk,
    const float* __restrict__ Wv, const float* __restrict__ bv,
    unsigned short* __restrict__ qo, unsigned short* __restrict__ ko,
    unsigned short* __restrict__ vo) {
    __shared__ unsigned short WT[DD * DD];   // transposed weight: WT[e*128 + d], bf16
    __shared__ unsigned short Xb[32 * DD];   // row-major features tile, bf16
    int tid = threadIdx.x, lane = tid & 31, wave = tid >> 5;
    int ntile = blockIdx.x;                  // 64 row tiles of 32
    int b = blockIdx.y / HH, h = blockIdx.y % HH;
    int sel = blockIdx.z;                    // 0=q 1=k 2=v
    const float* Wp = (sel == 0 ? Wq : sel == 1 ? Wk : Wv) + h * DD * DD;
    const float* bp = (sel == 0 ? bq : sel == 1 ? bk : bv) + h * DD;
    unsigned short* op = (sel == 0 ? qo : sel == 1 ? ko : vo);

    for (int i = tid * 4; i < DD * DD; i += 256 * 4) {       // stage W transposed as bf16
        float4 w = *(const float4*)(Wp + i);
        int d = i >> 7, e = i & 127;
        WT[(e + 0) * DD + d] = f2bf(w.x);
        WT[(e + 1) * DD + d] = f2bf(w.y);
        WT[(e + 2) * DD + d] = f2bf(w.z);
        WT[(e + 3) * DD + d] = f2bf(w.w);
    }
    const float* Xp = X + ((size_t)b * NN + (size_t)ntile * 32) * DD;
    for (int i = tid * 4; i < 32 * DD; i += 256 * 4) {       // stage X tile as bf16
        float4 x = *(const float4*)(Xp + i);
        uint2 pk;
        pk.x = (unsigned)f2bf(x.x) | ((unsigned)f2bf(x.y) << 16);
        pk.y = (unsigned)f2bf(x.z) | ((unsigned)f2bf(x.w) << 16);
        *(uint2*)&Xb[i] = pk;
    }
    __syncthreads();

    int rsub = wave & 1, tgrp = wave >> 1;
    int m = lane & 15, half = lane >> 4;
    for (int tt = 0; tt < 2; ++tt) {
        int col0 = (tgrp * 2 + tt) * 16;
        f32x8 acc = zero8();
        for (int kc = 0; kc < 4; ++kc) {
            Frag16 a = load_afrag(&Xb[rsub * 16 * DD + kc * 32], DD, lane);
            Frag16 bf = load_bfrag(&WT[col0 * DD + kc * 32], DD, lane);
            acc = wmma_bf16(a, bf, acc);
        }
        float bias = bp[col0 + m];
        size_t rowbase = (size_t)(b * HH + h) * NN + (size_t)ntile * 32 + rsub * 16;
        for (int r = 0; r < 8; ++r) {
            int n = r + 8 * half;                            // C-layout: row M = r + 8*half
            op[(rowbase + n) * DD + col0 + m] = f2bf(acc[r] + bias);
        }
    }
}

// ---------------- Kernel 3: flash attention with adjacency gating ----------------
__global__ __launch_bounds__(256) void attn_kernel(
    const unsigned short* __restrict__ qws, const unsigned short* __restrict__ kws,
    const unsigned short* __restrict__ vws, const float* __restrict__ asum,
    unsigned short* __restrict__ ows) {
    __shared__ unsigned short Qb[128 * DD];   // 32 KB: query rows of the block
    __shared__ unsigned short Kb[32 * DD];    // 8 KB: K tile, row-major (d contiguous = WMMA K-dim)
    __shared__ unsigned short VT[DD * 32];    // 8 KB: V tile transposed (m contiguous = WMMA K-dim)
    __shared__ unsigned short Pb[8][16 * 32]; // 8 KB: per-wave P scratch, C-layout -> A-layout
    int tid = threadIdx.x, lane = tid & 31, wave = tid >> 5;
    int n0 = blockIdx.x * 128, h = blockIdx.y, b = blockIdx.z;
    size_t headbase = (size_t)(b * HH + h) * NN * DD;
    int m = lane & 15, half = lane >> 4;

    {   // stage Q block (already bf16): 2048 uint4
        const uint4* src = (const uint4*)(qws + headbase + (size_t)n0 * DD);
        uint4* dst = (uint4*)Qb;
        for (int i = tid; i < 128 * DD / 8; i += 256) dst[i] = src[i];
    }
    __syncthreads();
    Frag16 qf[4];
    for (int kc = 0; kc < 4; ++kc)
        qf[kc] = load_afrag(&Qb[wave * 16 * DD + kc * 32], DD, lane);

    f32x8 o[8];
    for (int t = 0; t < 8; ++t) o[t] = zero8();
    float rm[8], rl[8];
    for (int r = 0; r < 8; ++r) { rm[r] = -3.0e38f; rl[r] = 0.0f; }
    const float scale = 0.02209708691207961f;                // 1/sqrt(2048)

    for (int m0 = 0; m0 < NN; m0 += 32) {
        __syncthreads();
        {   // K tile, row-major copy
            const uint4* src = (const uint4*)(kws + headbase + (size_t)m0 * DD);
            uint4* dst = (uint4*)Kb;
            for (int i = tid; i < 32 * DD / 8; i += 256) dst[i] = src[i];
        }
        {   // V tile, transposed store
            const uint4* src = (const uint4*)(vws + headbase + (size_t)m0 * DD);
            for (int i = tid; i < 32 * DD / 8; i += 256) {
                uint4 x = src[i];
                int ml = i >> 4, d0 = (i & 15) * 8;
                unsigned w[4] = {x.x, x.y, x.z, x.w};
                for (int j = 0; j < 4; ++j) {
                    VT[(d0 + 2 * j + 0) * 32 + ml] = (unsigned short)(w[j] & 0xFFFFu);
                    VT[(d0 + 2 * j + 1) * 32 + ml] = (unsigned short)(w[j] >> 16);
                }
            }
        }
        __syncthreads();

        // scores S = (q . k^T) * scale * adjsum, two 16x16 subtiles
        f32x8 s[2];
        for (int sub = 0; sub < 2; ++sub) {
            f32x8 acc = zero8();
            for (int kc = 0; kc < 4; ++kc) {
                Frag16 bf = load_bfrag(&Kb[(sub * 16) * DD + kc * 32], DD, lane);
                acc = wmma_bf16(qf[kc], bf, acc);
            }
            int mcol = m0 + sub * 16 + m;
            for (int r = 0; r < 8; ++r) {
                int ng = n0 + wave * 16 + r + 8 * half;
                acc[r] *= scale * asum[((size_t)b * NN + ng) * NN + mcol];
            }
            s[sub] = acc;
        }

        // online softmax: row stats live per-lane aligned with the C-layout rows
        for (int r = 0; r < 8; ++r) {
            float t0 = fmaxf(s[0][r], s[1][r]);
            for (int off = 1; off < 16; off <<= 1) t0 = fmaxf(t0, __shfl_xor(t0, off, 32));
            float nm = fmaxf(rm[r], t0);
            float f = __expf(rm[r] - nm);
            float p0 = __expf(s[0][r] - nm);
            float p1 = __expf(s[1][r] - nm);
            float rs = p0 + p1;
            for (int off = 1; off < 16; off <<= 1) rs += __shfl_xor(rs, off, 32);
            rl[r] = rl[r] * f + rs;
            rm[r] = nm;
            for (int t = 0; t < 8; ++t) o[t][r] *= f;
            int row = r + 8 * half;
            Pb[wave][row * 32 + m]      = f2bf(p0);
            Pb[wave][row * 32 + 16 + m] = f2bf(p1);
        }
        asm volatile("s_wait_dscnt 0x0" ::: "memory");       // P writes -> frag reads (same wave)

        Frag16 pf = load_afrag(&Pb[wave][0], 32, lane);       // one 16x32 A-frag covers the tile
        for (int t = 0; t < 8; ++t) {
            Frag16 vf = load_bfrag(&VT[(t * 16) * 32], 32, lane);
            o[t] = wmma_bf16(pf, vf, o[t]);
        }
    }

    for (int r = 0; r < 8; ++r) {                             // normalize + concat-heads store
        float inv = 1.0f / rl[r];
        int ng = n0 + wave * 16 + r + 8 * half;
        size_t base = ((size_t)b * NN + ng) * (HH * DD) + h * DD;
        for (int t = 0; t < 8; ++t) ows[base + t * 16 + m] = f2bf(o[t][r] * inv);
    }
}

// ---------------- Kernel 4: output GEMM [4096x1024]x[1024x256] + bias ----------------
__global__ __launch_bounds__(256) void out_kernel(
    const unsigned short* __restrict__ ows, const float* __restrict__ Wout,
    const float* __restrict__ bout, float* __restrict__ out) {
    __shared__ unsigned short Ab[32 * 32];     // 2 KB
    __shared__ unsigned short WTt[OUTD * 32];  // 16 KB, transposed: WTt[c*32 + k]
    int tid = threadIdx.x, lane = tid & 31, wave = tid >> 5;
    int row0 = blockIdx.x * 32;
    int m = lane & 15, half = lane >> 4;
    int rsub = wave & 1;
    f32x8 acc[4];
    for (int t = 0; t < 4; ++t) acc[t] = zero8();

    for (int k0 = 0; k0 < HH * DD; k0 += 32) {
        __syncthreads();
        for (int i = tid; i < 128; i += 256) {               // A tile 32x32 bf16
            int r = i >> 2, c = (i & 3) * 8;
            *(uint4*)&Ab[r * 32 + c] =
                *(const uint4*)(ows + (size_t)(row0 + r) * (HH * DD) + k0 + c);
        }
        for (int i = tid * 4; i < 32 * OUTD; i += 256 * 4) { // W tile transposed -> bf16
            int kr = i >> 8, c = i & 255;
            float4 w = *(const float4*)(Wout + (size_t)(k0 + kr) * OUTD + c);
            WTt[(c + 0) * 32 + kr] = f2bf(w.x);
            WTt[(c + 1) * 32 + kr] = f2bf(w.y);
            WTt[(c + 2) * 32 + kr] = f2bf(w.z);
            WTt[(c + 3) * 32 + kr] = f2bf(w.w);
        }
        __syncthreads();
        Frag16 a = load_afrag(&Ab[rsub * 16 * 32], 32, lane);
        for (int t = 0; t < 4; ++t) {
            int tile = (wave >> 1) * 4 + t;
            Frag16 bf = load_bfrag(&WTt[tile * 16 * 32], 32, lane);
            acc[t] = wmma_bf16(a, bf, acc[t]);
        }
    }
    for (int t = 0; t < 4; ++t) {
        int col = ((wave >> 1) * 4 + t) * 16 + m;
        float bb = bout[col];
        for (int r = 0; r < 8; ++r) {
            int rowg = row0 + rsub * 16 + r + 8 * half;
            out[(size_t)rowg * OUTD + col] = acc[t][r] + bb;
        }
    }
}

extern "C" void kernel_launch(void* const* d_in, const int* in_sizes, int n_in,
                              void* d_out, int out_size, void* d_ws, size_t ws_size,
                              hipStream_t stream) {
    const float* X    = (const float*)d_in[0];
    const float* adj  = (const float*)d_in[1];
    const float* Wq   = (const float*)d_in[2];
    const float* bq   = (const float*)d_in[3];
    const float* Wk   = (const float*)d_in[4];
    const float* bk   = (const float*)d_in[5];
    const float* Wv   = (const float*)d_in[6];
    const float* bv   = (const float*)d_in[7];
    const float* Wout = (const float*)d_in[8];
    const float* bout = (const float*)d_in[9];
    float* out = (float*)d_out;

    char* ws = (char*)d_ws;
    size_t qkv_elems = (size_t)BB * HH * NN * DD;            // 4 Mi elems
    unsigned short* qws = (unsigned short*)ws; ws += qkv_elems * 2;   // 8 MB
    unsigned short* kws = (unsigned short*)ws; ws += qkv_elems * 2;   // 8 MB
    unsigned short* vws = (unsigned short*)ws; ws += qkv_elems * 2;   // 8 MB
    float* asum = (float*)ws;                  ws += (size_t)BB * NN * NN * 4;  // 32 MB
    unsigned short* ows = (unsigned short*)ws;                        // 8 MB

    long long nadj = (long long)BB * NN * NN;
    adjsum_kernel<<<dim3((unsigned)((nadj + 255) / 256)), 256, 0, stream>>>(
        (const float4*)adj, asum, nadj);
    qkv_kernel<<<dim3(NN / 32, BB * HH, 3), 256, 0, stream>>>(
        X, Wq, bq, Wk, bk, Wv, bv, qws, kws, vws);
    attn_kernel<<<dim3(NN / 128, HH, BB), 256, 0, stream>>>(qws, kws, vws, asum, ows);
    out_kernel<<<dim3(BB * NN / 32), 256, 0, stream>>>(ows, Wout, bout, out);
}
